// GPTDecoderLayer_7593502179339
// MI455X (gfx1250) — compile-verified, matrix operand order checked by reading, not run on hardware
//
#include <hip/hip_runtime.h>

// ---------------- problem constants ----------------
#define B_   2
#define T_   2048
#define D_   1024
#define H_   16
#define KS_  64
#define ROT_ 32
#define FF_  4096
#define EPS_ 1e-6f
#define BT_  (B_ * T_)          // 4096 rows

typedef __bf16 bf16_t;
typedef __attribute__((ext_vector_type(16))) __bf16 v16bf;
typedef __attribute__((ext_vector_type(8)))  float  v8f;
typedef __attribute__((ext_vector_type(4)))  unsigned u32x4;
typedef __attribute__((ext_vector_type(8)))  unsigned u32x8;

union FragBF {                   // one WMMA 16x32 bf16 operand (8 VGPRs)
    uint4 u[2];
    v16bf v;
};

__device__ __forceinline__ bf16_t f2bf(float f) {
    union { float f; unsigned u; } in;
    in.f = f;
    unsigned r = in.u + 0x7FFFu + ((in.u >> 16) & 1u);   // round-to-nearest-even
    unsigned short hs = (unsigned short)(r >> 16);
    union { unsigned short u; bf16_t b; } out;
    out.u = hs;
    return out.b;
}

// --- CDNA5 async Global->LDS copy (ASYNCcnt-tracked, bypasses VGPRs) ---
__device__ __forceinline__ unsigned lds_off(const void* p) {
    return (unsigned)(uintptr_t)p;   // low 32 bits of flat-shared addr = LDS byte address
}
__device__ __forceinline__ void async_copy_b128(unsigned lds_addr, const void* gaddr) {
    asm volatile("global_load_async_to_lds_b128 %0, %1, off"
                 :: "v"(lds_addr), "v"(gaddr) : "memory");
}
__device__ __forceinline__ void wait_async0() {
    asm volatile("s_wait_asynccnt 0" ::: "memory");
}

// --- CDNA5 Tensor Data Mover: 2D tile Global->LDS (TENSORcnt-tracked) ---
// D# per ISA 8.3/8.4: group0 = {flags, lds_addr, global_addr, type=2},
// group1 = {mask/data_size/pad, tensor dims, tile dims, dim0 stride}.
__device__ __forceinline__ void tdm_load_2d(unsigned lds_addr, const void* gaddr,
                                            unsigned tile_d0, unsigned tile_d1,
                                            unsigned tensor_d0, unsigned tensor_d1,
                                            unsigned stride_d0_elems,
                                            unsigned pad_interval, unsigned pad_amount) {
    unsigned long long ga = (unsigned long long)(uintptr_t)gaddr;
    u32x4 g0;
    g0.x = 1u;                                             // count=1, user descriptor
    g0.y = lds_addr;                                       // LDS byte address
    g0.z = (unsigned)(ga & 0xFFFFFFFFu);                   // global_addr[31:0]
    g0.w = (unsigned)((ga >> 32) & 0x01FFFFFFu) | (2u << 30);  // addr[56:32] | type=2
    u32x8 g1;
    g1.s0 = (1u << 16)                                     // data_size = 2 bytes
          | (1u << 20)                                     // pad_enable
          | (pad_interval << 22) | (pad_amount << 25);
    g1.s1 = (tensor_d0 & 0xFFFFu) << 16;                   // tensor_dim0[15:0] -> bits63:48
    g1.s2 = ((tensor_d0 >> 16) & 0xFFFFu) | ((tensor_d1 & 0xFFFFu) << 16);
    g1.s3 = ((tensor_d1 >> 16) & 0xFFFFu) | ((tile_d0 & 0xFFFFu) << 16);
    g1.s4 = (tile_d1 & 0xFFFFu);                           // tile_dim1; tile_dim2 = 0
    g1.s5 = stride_d0_elems;                               // tensor_dim0_stride[31:0]
    g1.s6 = 0u;                                            // stride hi / dim1 stride lo
    g1.s7 = 0u;
    asm volatile("tensor_load_to_lds %0, %1, null, null"
                 :: "s"(g0), "s"(g1) : "memory");
}
__device__ __forceinline__ void wait_tensor0() {
    asm volatile("s_wait_tensorcnt 0" ::: "memory");
}

// ---------------- rms_norm: fp32 in -> bf16 out ----------------
__global__ __launch_bounds__(256) void rmsnorm_kernel(const float* __restrict__ x,
                                                      const float* __restrict__ scale,
                                                      bf16_t* __restrict__ out) {
    int row = blockIdx.x;                    // 0..BT-1
    int tid = threadIdx.x;
    const float* xr = x + (size_t)row * D_;
    float vals[4];
    float ss = 0.f;
#pragma unroll
    for (int i = 0; i < 4; ++i) {
        vals[i] = xr[tid + i * 256];
        ss += vals[i] * vals[i];
    }
#pragma unroll
    for (int m = 16; m >= 1; m >>= 1) ss += __shfl_xor(ss, m, 32);
    __shared__ float red[8];
    if ((tid & 31) == 0) red[tid >> 5] = ss;
    __syncthreads();
    float tot = red[0] + red[1] + red[2] + red[3] + red[4] + red[5] + red[6] + red[7];
    float r = rsqrtf(tot / (float)D_ + EPS_);
    bf16_t* o = out + (size_t)row * D_;
#pragma unroll
    for (int i = 0; i < 4; ++i)
        o[tid + i * 256] = f2bf(vals[i] * scale[tid + i * 256] * r);
}

// ---------------- fp32 -> bf16 weight convert ----------------
__global__ __launch_bounds__(256) void cvt_bf16_kernel(const float* __restrict__ in,
                                                       bf16_t* __restrict__ out, int n) {
    int i = blockIdx.x * 256 + threadIdx.x;
    if (i < n) out[i] = f2bf(in[i]);
}

// ---------------- tiled bf16 WMMA GEMM, double-buffered TDM pipeline ----------------
// C[M,N] = A[M,K] @ Bw[K,N]; block tile 128x128, K-step 32, 8 waves of 32.
enum { EPI_ROPE = 0, EPI_VT = 1, EPI_RES = 2, EPI_SWIGLU = 3 };

template <int EPI, bool DUAL>
__global__ __launch_bounds__(256) void gemm_bf16_kernel(
    const bf16_t* __restrict__ A, const bf16_t* __restrict__ Bw,
    const float* __restrict__ res, float* __restrict__ outf, bf16_t* __restrict__ outb,
    int M, int N, int K, int ldb) {
    constexpr int LDA = 48;   // A pitch: 32 + 16 pad (matches TDM pad 8 DWORDs)
    constexpr int LDB = 32;   // B^T pitch: rows stay 16B aligned
    __shared__ alignas(16) bf16_t sA[2][128 * LDA];
    __shared__ alignas(16) bf16_t sB[2][128 * LDB];
    __shared__ alignas(16) bf16_t sB2[DUAL ? 2 : 1][DUAL ? 128 * LDB : 16];

    const int tid  = threadIdx.x;
    const int lane = tid & 31;
    const int wave = tid >> 5;
    const int wm = wave >> 2;         // 0..1  (row half of block)
    const int wn = wave & 3;          // 0..3  (col quarter of block)
    const int hf = lane >> 4;         // half-wave
    const int tlo = lane & 15;

    const int rowB = blockIdx.y * 128;
    const int colB = blockIdx.x * 128;

    v8f acc[4][2];
    v8f acc2[4][2];
#pragma unroll
    for (int i = 0; i < 4; ++i)
#pragma unroll
        for (int j = 0; j < 2; ++j) { acc[i][j] = {}; acc2[i][j] = {}; }

    // stage tile (A via TDM from wave 0; B manual transpose) into buffer bufi
    auto stage = [&](int k0s, int bufi) {
        if (wave == 0)
            tdm_load_2d(lds_off(&sA[bufi][0]), A + (size_t)rowB * K + k0s,
                        /*tile*/32, 128, /*tensor*/32, 128,
                        /*stride*/(unsigned)K,
                        /*pad_interval: 16 DW rows*/3, /*pad: 8 DW*/7);
#pragma unroll
        for (int p = 0; p < 2; ++p) {
            int kr = p * 16 + (tid >> 4);
            int n8 = (tid & 15) * 8;
            uint4 vv = *(const uint4*)(Bw + (size_t)(k0s + kr) * ldb + colB + n8);
            const bf16_t* hsrc = (const bf16_t*)&vv;
#pragma unroll
            for (int j = 0; j < 8; ++j) sB[bufi][(n8 + j) * LDB + kr] = hsrc[j];
            if (DUAL) {
                uint4 v2 = *(const uint4*)(Bw + (size_t)(k0s + kr) * ldb + colB + FF_ + n8);
                const bf16_t* hs2 = (const bf16_t*)&v2;
#pragma unroll
                for (int j = 0; j < 8; ++j)
                    sB2[DUAL ? bufi : 0][(n8 + j) * LDB + kr] = hs2[j];
            }
        }
    };

    stage(0, 0);
    if (wave == 0) wait_tensor0();
    __syncthreads();

    for (int k0 = 0; k0 < K; k0 += 32) {
        const int cur = (k0 >> 5) & 1;
        const bool more = (k0 + 32) < K;
        if (more) stage(k0 + 32, cur ^ 1);   // overlap next-tile DMA with compute

        FragBF fa[4], fb[2], fb2[2];
#pragma unroll
        for (int mt = 0; mt < 4; ++mt) {
            const bf16_t* p = &sA[cur][(wm * 64 + mt * 16 + tlo) * LDA + hf * 8];
            fa[mt].u[0] = *(const uint4*)p;
            fa[mt].u[1] = *(const uint4*)(p + 16);
        }
#pragma unroll
        for (int nt = 0; nt < 2; ++nt) {
            const bf16_t* p = &sB[cur][(wn * 32 + nt * 16 + tlo) * LDB + hf * 8];
            fb[nt].u[0] = *(const uint4*)p;
            fb[nt].u[1] = *(const uint4*)(p + 16);
            if (DUAL) {
                const bf16_t* q = &sB2[DUAL ? cur : 0][(wn * 32 + nt * 16 + tlo) * LDB + hf * 8];
                fb2[nt].u[0] = *(const uint4*)q;
                fb2[nt].u[1] = *(const uint4*)(q + 16);
            }
        }
#pragma unroll
        for (int mt = 0; mt < 4; ++mt)
#pragma unroll
            for (int nt = 0; nt < 2; ++nt) {
                acc[mt][nt] = __builtin_amdgcn_wmma_f32_16x16x32_bf16(
                    false, fa[mt].v, false, fb[nt].v, (short)0, acc[mt][nt], false, false);
                if (DUAL)
                    acc2[mt][nt] = __builtin_amdgcn_wmma_f32_16x16x32_bf16(
                        false, fa[mt].v, false, fb2[nt].v, (short)0, acc2[mt][nt], false, false);
            }

        if (more) {
            if (wave == 0) wait_tensor0();   // next A tile landed
            __syncthreads();                 // publishes next tile, retires this one
        }
    }

    // --- epilogue ---
    const int c0 = colB + wn * 32;
    const int r0 = rowB + wm * 64;
#pragma unroll
    for (int mt = 0; mt < 4; ++mt)
#pragma unroll
        for (int nt = 0; nt < 2; ++nt) {
            int col = c0 + nt * 16 + tlo;
#pragma unroll
            for (int r = 0; r < 8; ++r) {
                int row = r0 + mt * 16 + hf * 8 + r;
                float v = acc[mt][nt][r];
                float partner = __shfl_xor(v, 1, 32);   // rotate-every-two neighbor
                if (EPI == EPI_RES) {
                    outf[(size_t)row * N + col] = v + res[(size_t)row * N + col];
                } else if (EPI == EPI_ROPE) {
                    int t = row & (T_ - 1);
                    int b = row >> 11;               // T_ = 2048
                    int h = col >> 6;                // KS_ = 64
                    int d = col & 63;
                    float vo = v;
                    if (d < ROT_) {
                        int i = d >> 1;
                        float ang = (float)t * __powf(10000.0f, -(float)(2 * i) / (float)ROT_);
                        float s, c;
                        __sincosf(ang, &s, &c);
                        vo = (d & 1) ? (v * c + partner * s) : (v * c - partner * s);
                    }
                    outb[(((size_t)(b * H_ + h)) * T_ + t) * KS_ + d] = f2bf(vo);
                } else if (EPI == EPI_VT) {
                    int t = row & (T_ - 1);
                    int b = row >> 11;
                    int h = col >> 6;
                    int d = col & 63;
                    outb[(((size_t)(b * H_ + h)) * KS_ + d) * T_ + t] = f2bf(v);
                } else { // EPI_SWIGLU
                    float x1 = v;
                    float x2 = acc2[mt][nt][r];
                    float g = (x1 / (1.0f + __expf(-x1))) * x2;
                    outb[(size_t)row * N + col] = f2bf(g);
                }
            }
        }
}

// ---------------- flash attention (causal), bf16 WMMA, f32 softmax ----------------
// q,k: [B,H,T,KS] bf16 (RoPE applied); vt: [B,H,KS,T] bf16; out: [B*T, D] bf16
__global__ __launch_bounds__(128) void flash_kernel(const bf16_t* __restrict__ q,
                                                    const bf16_t* __restrict__ k,
                                                    const bf16_t* __restrict__ vt,
                                                    bf16_t* __restrict__ attn_out) {
    constexpr int LKV = 80;    // LDS pitch (elems); 160B rows -> 16B aligned
    __shared__ alignas(16) bf16_t sK[64 * LKV];
    __shared__ alignas(16) bf16_t sV[64 * LKV];
    __shared__ alignas(16) bf16_t sP[4][16 * LKV];

    const int bh = blockIdx.y;           // b*H + h
    const int rb = blockIdx.x;           // 64-row block
    const int tid = threadIdx.x;
    const int lane = tid & 31;
    const int wave = tid >> 5;
    const int hf = lane >> 4;
    const int tlo = lane & 15;

    const bf16_t* qbh = q + (size_t)bh * T_ * KS_;
    const bf16_t* kbh = k + (size_t)bh * T_ * KS_;
    const bf16_t* vbh = vt + (size_t)bh * KS_ * T_;

    const int row0 = rb * 64 + wave * 16;

    // Q fragments for this wave's 16 rows (d = 0..63 -> 2 K-steps)
    FragBF aq[2];
    {
        const bf16_t* qp = qbh + (size_t)(row0 + tlo) * KS_;
#pragma unroll
        for (int ks = 0; ks < 2; ++ks) {
            aq[ks].u[0] = *(const uint4*)(qp + ks * 32 + hf * 8);
            aq[ks].u[1] = *(const uint4*)(qp + ks * 32 + hf * 8 + 16);
        }
    }

    float mrow[8], lrow[8];
    v8f acco[4];
#pragma unroll
    for (int r = 0; r < 8; ++r) { mrow[r] = -3.0e38f; lrow[r] = 0.f; }
#pragma unroll
    for (int d = 0; d < 4; ++d) acco[d] = {};

    const int nkb = rb + 1;               // causal: key blocks 0..rb
    for (int kb = 0; kb < nkb; ++kb) {
        const int key0 = kb * 64;
        __syncthreads();
        {   // K/V tiles via async Global->LDS (ASYNCcnt), 32 elems/thread/tile
            int r = tid >> 1;
            int c = (tid & 1) * 32;
            const bf16_t* ksrc = kbh + (size_t)(key0 + r) * KS_ + c;
            const bf16_t* vsrc = vbh + (size_t)r * T_ + key0 + c;
#pragma unroll
            for (int j = 0; j < 4; ++j) {
                async_copy_b128(lds_off(&sK[r * LKV + c + j * 8]), ksrc + j * 8);
                async_copy_b128(lds_off(&sV[r * LKV + c + j * 8]), vsrc + j * 8);
            }
        }
        wait_async0();
        __syncthreads();

        // S = Q @ K^T for 16 rows x 64 keys
        v8f accs[4];
#pragma unroll
        for (int nt = 0; nt < 4; ++nt) accs[nt] = {};
#pragma unroll
        for (int ks = 0; ks < 2; ++ks) {
#pragma unroll
            for (int nt = 0; nt < 4; ++nt) {
                FragBF bk;
                const bf16_t* p = &sK[(nt * 16 + tlo) * LKV + ks * 32 + hf * 8];
                bk.u[0] = *(const uint4*)p;
                bk.u[1] = *(const uint4*)(p + 16);
                accs[nt] = __builtin_amdgcn_wmma_f32_16x16x32_bf16(
                    false, aq[ks].v, false, bk.v, (short)0, accs[nt], false, false);
            }
        }

        // online softmax (rows live in 16 lanes of one half-wave)
        float mnew[8];
#pragma unroll
        for (int r = 0; r < 8; ++r) {
            int rowg = row0 + hf * 8 + r;
            float rmax = -3.0e38f;
#pragma unroll
            for (int nt = 0; nt < 4; ++nt) {
                int key = key0 + nt * 16 + tlo;
                float sv = accs[nt][r] * 0.125f;         // 1/sqrt(64)
                sv = (key <= rowg) ? sv : -1.0e30f;       // causal mask
                accs[nt][r] = sv;
                rmax = fmaxf(rmax, sv);
            }
#pragma unroll
            for (int m = 1; m <= 8; m <<= 1) rmax = fmaxf(rmax, __shfl_xor(rmax, m, 32));
            mnew[r] = fmaxf(mrow[r], rmax);
        }
#pragma unroll
        for (int r = 0; r < 8; ++r) {
            float alpha = __expf(mrow[r] - mnew[r]);
            float rsum = 0.f;
#pragma unroll
            for (int nt = 0; nt < 4; ++nt) {
                float p = __expf(accs[nt][r] - mnew[r]);
                accs[nt][r] = p;
                rsum += p;
            }
#pragma unroll
            for (int m = 1; m <= 8; m <<= 1) rsum += __shfl_xor(rsum, m, 32);
            lrow[r] = lrow[r] * alpha + rsum;
            mrow[r] = mnew[r];
#pragma unroll
            for (int d = 0; d < 4; ++d) acco[d][r] *= alpha;
        }

        // stage P (bf16) to per-wave LDS strip, then rebuild as A fragments
        bf16_t* pw = &sP[wave][0];
#pragma unroll
        for (int nt = 0; nt < 4; ++nt)
#pragma unroll
            for (int r = 0; r < 8; ++r)
                pw[(hf * 8 + r) * LKV + nt * 16 + tlo] = f2bf(accs[nt][r]);
        asm volatile("s_wait_dscnt 0" ::: "memory");   // wave-local store->load ordering

        FragBF ap[2];
#pragma unroll
        for (int ks = 0; ks < 2; ++ks) {
            const bf16_t* p = &pw[tlo * LKV + ks * 32 + hf * 8];
            ap[ks].u[0] = *(const uint4*)p;
            ap[ks].u[1] = *(const uint4*)(p + 16);
        }
        // O += P @ V
#pragma unroll
        for (int ks = 0; ks < 2; ++ks) {
#pragma unroll
            for (int dt = 0; dt < 4; ++dt) {
                FragBF bv;
                const bf16_t* p = &sV[(dt * 16 + tlo) * LKV + ks * 32 + hf * 8];
                bv.u[0] = *(const uint4*)p;
                bv.u[1] = *(const uint4*)(p + 16);
                acco[dt] = __builtin_amdgcn_wmma_f32_16x16x32_bf16(
                    false, ap[ks].v, false, bv.v, (short)0, acco[dt], false, false);
            }
        }
    }

    // normalize and write out: [b*T + t][h*KS + d]
    const int b = bh >> 4;     // H_ = 16
    const int h = bh & 15;
#pragma unroll
    for (int dt = 0; dt < 4; ++dt)
#pragma unroll
        for (int r = 0; r < 8; ++r) {
            int rowg = row0 + hf * 8 + r;
            float o = acco[dt][r] / lrow[r];
            int d = dt * 16 + tlo;
            attn_out[((size_t)(b * T_ + rowg)) * D_ + h * KS_ + d] = f2bf(o);
        }
}

// ---------------- host orchestration ----------------
extern "C" void kernel_launch(void* const* d_in, const int* in_sizes, int n_in,
                              void* d_out, int out_size, void* d_ws, size_t ws_size,
                              hipStream_t stream) {
    const float* x      = (const float*)d_in[0];
    // d_in[1] = attention_mask (causal; computed analytically)
    const float* Wq     = (const float*)d_in[2];
    const float* Wk     = (const float*)d_in[3];
    const float* Wv     = (const float*)d_in[4];
    const float* Wo     = (const float*)d_in[5];
    const float* ascale = (const float*)d_in[6];
    const float* fscale = (const float*)d_in[7];
    const float* W1     = (const float*)d_in[8];
    const float* W2     = (const float*)d_in[9];
    float* out = (float*)d_out;

    char* ws = (char*)d_ws;
    auto alloc = [&](size_t bytes) -> char* {
        char* p = ws;
        ws += (bytes + 255) & ~(size_t)255;
        return p;
    };
    bf16_t* xn    = (bf16_t*)alloc((size_t)BT_ * D_ * 2);
    bf16_t* wq_b  = (bf16_t*)alloc((size_t)D_ * D_ * 2);
    bf16_t* wk_b  = (bf16_t*)alloc((size_t)D_ * D_ * 2);
    bf16_t* wv_b  = (bf16_t*)alloc((size_t)D_ * D_ * 2);
    bf16_t* wo_b  = (bf16_t*)alloc((size_t)D_ * D_ * 2);
    bf16_t* w1_b  = (bf16_t*)alloc((size_t)D_ * 2 * FF_ * 2);
    bf16_t* w2_b  = (bf16_t*)alloc((size_t)FF_ * D_ * 2);
    bf16_t* q_b   = (bf16_t*)alloc((size_t)BT_ * D_ * 2);
    bf16_t* k_b   = (bf16_t*)alloc((size_t)BT_ * D_ * 2);
    bf16_t* vt_b  = (bf16_t*)alloc((size_t)BT_ * D_ * 2);
    bf16_t* at_b  = (bf16_t*)alloc((size_t)BT_ * D_ * 2);
    float*  hbuf  = (float*) alloc((size_t)BT_ * D_ * 4);
    bf16_t* hn    = (bf16_t*)alloc((size_t)BT_ * D_ * 2);
    bf16_t* g_b   = (bf16_t*)alloc((size_t)BT_ * FF_ * 2);

    // weight converts (fp32 -> bf16)
    cvt_bf16_kernel<<<(D_ * D_) / 256, 256, 0, stream>>>(Wq, wq_b, D_ * D_);
    cvt_bf16_kernel<<<(D_ * D_) / 256, 256, 0, stream>>>(Wk, wk_b, D_ * D_);
    cvt_bf16_kernel<<<(D_ * D_) / 256, 256, 0, stream>>>(Wv, wv_b, D_ * D_);
    cvt_bf16_kernel<<<(D_ * D_) / 256, 256, 0, stream>>>(Wo, wo_b, D_ * D_);
    cvt_bf16_kernel<<<(D_ * 2 * FF_) / 256, 256, 0, stream>>>(W1, w1_b, D_ * 2 * FF_);
    cvt_bf16_kernel<<<(FF_ * D_) / 256, 256, 0, stream>>>(W2, w2_b, FF_ * D_);

    // xn = rms_norm(x, attn_scale)
    rmsnorm_kernel<<<BT_, 256, 0, stream>>>(x, ascale, xn);

    // Q/K with fused RoPE -> [B,H,T,KS]; V transposed -> [B,H,KS,T]
    dim3 gemm_blk(256);
    dim3 grid_qkv(D_ / 128, BT_ / 128);
    gemm_bf16_kernel<EPI_ROPE, false><<<grid_qkv, gemm_blk, 0, stream>>>(
        xn, wq_b, nullptr, nullptr, q_b, BT_, D_, D_, D_);
    gemm_bf16_kernel<EPI_ROPE, false><<<grid_qkv, gemm_blk, 0, stream>>>(
        xn, wk_b, nullptr, nullptr, k_b, BT_, D_, D_, D_);
    gemm_bf16_kernel<EPI_VT, false><<<grid_qkv, gemm_blk, 0, stream>>>(
        xn, wv_b, nullptr, nullptr, vt_b, BT_, D_, D_, D_);

    // flash attention
    dim3 fa_grid(T_ / 64, B_ * H_);
    flash_kernel<<<fa_grid, 128, 0, stream>>>(q_b, k_b, vt_b, at_b);

    // h = x + attn @ Wo
    gemm_bf16_kernel<EPI_RES, false><<<dim3(D_ / 128, BT_ / 128), gemm_blk, 0, stream>>>(
        at_b, wo_b, x, hbuf, nullptr, BT_, D_, D_, D_);

    // hn = rms_norm(h, ffn_scale)
    rmsnorm_kernel<<<BT_, 256, 0, stream>>>(hbuf, fscale, hn);

    // g = swish(hn @ W1[:, :FF]) * (hn @ W1[:, FF:])
    gemm_bf16_kernel<EPI_SWIGLU, true><<<dim3(FF_ / 128, BT_ / 128), gemm_blk, 0, stream>>>(
        hn, w1_b, nullptr, nullptr, g_b, BT_, FF_, D_, 2 * FF_);

    // out = h + g @ W2
    gemm_bf16_kernel<EPI_RES, false><<<dim3(D_ / 128, BT_ / 128), gemm_blk, 0, stream>>>(
        g_b, w2_b, hbuf, out, nullptr, BT_, D_, FF_, D_);
}
